// MambaBlock_32452772889243
// MI455X (gfx1250) — matrix operand order repaired and checked by reading, not compiled, not run on hardware
//
#include <hip/hip_runtime.h>

// ---------------------------------------------------------------------------
// MambaBlock forward for MI455X (gfx1250): bf16 WMMA GEMMs with TDM-pipelined
// LDS staging + wave32 selective scan.
// B=4, L=2048, D=1024, N=16, Dff=3072, T = B*L = 8192 tokens.
// ---------------------------------------------------------------------------

#define BB 4
#define LL 2048
#define DM 1024
#define NS 16
#define DFF 3072
#define TT (BB * LL)          // 8192 tokens

typedef __attribute__((ext_vector_type(16))) __bf16 v16bf;
typedef __attribute__((ext_vector_type(8)))  float  v8f;

#if defined(__HIP_DEVICE_COMPILE__)
#if __has_builtin(__builtin_amdgcn_tensor_load_to_lds)
#define USE_TDM 1
#endif
#endif

__device__ __forceinline__ unsigned short f2bf(float f) {
    unsigned int u = __float_as_uint(f);
    unsigned int r = (u + 0x7FFFu + ((u >> 16) & 1u)) >> 16;   // RNE
    return (unsigned short)r;
}

#ifdef USE_TDM
typedef unsigned int v4u_t __attribute__((ext_vector_type(4)));
typedef int v8i_t __attribute__((ext_vector_type(8)));
typedef int v4i_t __attribute__((ext_vector_type(4)));

// Issue a TDM 2D tile load (bf16 elements): tile (tile1 rows x tile0 cols)
// from a row-major (td1 x td0) tensor with row stride stride0 (elements),
// packed row-major & contiguous into LDS at byte offset lds_off.
__device__ __forceinline__ void tdm_load_tile_bf16(
    const unsigned short* g, unsigned lds_off,
    unsigned td0, unsigned td1, unsigned tile0, unsigned tile1,
    unsigned stride0) {
    unsigned long long ga = (unsigned long long)(uintptr_t)g;
    v4u_t g0;
    g0[0] = 1u;                                        // count=1 (valid D#)
    g0[1] = lds_off;                                   // lds_addr
    g0[2] = (unsigned)ga;                              // global_addr[31:0]
    g0[3] = (unsigned)((ga >> 32) & 0x01FFFFFFu) | (2u << 30); // addr[56:32]|type=2
    v8i_t g1;
    g1[0] = (int)(1u << 16);                           // data_size=1 -> 2 bytes
    g1[1] = (int)((td0 & 0xFFFFu) << 16);              // tensor_dim0[15:0]
    g1[2] = (int)(((td0 >> 16) & 0xFFFFu) | ((td1 & 0xFFFFu) << 16));
    g1[3] = (int)(((td1 >> 16) & 0xFFFFu) | ((tile0 & 0xFFFFu) << 16));
    g1[4] = (int)(tile1 & 0xFFFFu);                    // tile_dim1 (tile_dim2=0)
    g1[5] = (int)stride0;                              // tensor_dim0_stride[31:0]
    g1[6] = 0;
    g1[7] = 0;
    v4i_t z = {0, 0, 0, 0};
#if __clang_major__ >= 23
    v8i_t z8 = {0, 0, 0, 0, 0, 0, 0, 0};
    __builtin_amdgcn_tensor_load_to_lds(g0, g1, z, z, z8, 0);
#else
    __builtin_amdgcn_tensor_load_to_lds(g0, g1, z, z, 0);
#endif
}
#endif

// ---------------------------------------------------------------------------
// f32 -> bf16 bulk convert
// ---------------------------------------------------------------------------
__global__ __launch_bounds__(256) void cvt_bf16_kernel(
    const float* __restrict__ src, unsigned short* __restrict__ dst, int n) {
    int i = blockIdx.x * 256 + threadIdx.x;
    if (i < n) dst[i] = f2bf(src[i]);
}

// ---------------------------------------------------------------------------
// Build padded [Bp | Cp | 0] weight (1024 x 64, bf16) and bias (64)
// ---------------------------------------------------------------------------
__global__ __launch_bounds__(256) void pack_bc_kernel(
    const float* __restrict__ bpw, const float* __restrict__ cpw,
    const float* __restrict__ bpb, const float* __restrict__ cpb,
    unsigned short* __restrict__ wbc, float* __restrict__ bcat) {
    int i = blockIdx.x * 256 + threadIdx.x;           // over 1024*64
    if (i >= DM * 64) return;
    int k = i >> 6, c = i & 63;
    float v = 0.f;
    if (c < 16)       v = bpw[k * NS + c];
    else if (c < 32)  v = cpw[k * NS + (c - 16)];
    wbc[i] = f2bf(v);
    if (k == 0) bcat[c] = (c < 16) ? bpb[c] : (c < 32 ? cpb[c - 16] : 0.f);
}

// ---------------------------------------------------------------------------
// LayerNorm (1 or 2 chained norms), D=1024, one block (256 thr) per token.
// ---------------------------------------------------------------------------
template <int NLN>
__global__ __launch_bounds__(256) void ln_kernel(
    const float* __restrict__ x,
    const float* __restrict__ g1, const float* __restrict__ b1,
    const float* __restrict__ g2, const float* __restrict__ b2,
    unsigned short* __restrict__ out) {
    __shared__ float s1[256], s2[256];
    const int row = blockIdx.x, tid = threadIdx.x;
    const float* xr = x + (size_t)row * DM;
    float4 v = reinterpret_cast<const float4*>(xr)[tid];
    s1[tid] = v.x + v.y + v.z + v.w;
    s2[tid] = v.x * v.x + v.y * v.y + v.z * v.z + v.w * v.w;
    __syncthreads();
    for (int st = 128; st > 0; st >>= 1) {
        if (tid < st) { s1[tid] += s1[tid + st]; s2[tid] += s2[tid + st]; }
        __syncthreads();
    }
    float mu = s1[0] * (1.f / DM);
    float var = s2[0] * (1.f / DM) - mu * mu;
    float rs = rsqrtf(var + 1e-5f);
    __syncthreads();
    float4 g = reinterpret_cast<const float4*>(g1)[tid];
    float4 bb = reinterpret_cast<const float4*>(b1)[tid];
    float y0 = (v.x - mu) * rs * g.x + bb.x;
    float y1 = (v.y - mu) * rs * g.y + bb.y;
    float y2 = (v.z - mu) * rs * g.z + bb.z;
    float y3 = (v.w - mu) * rs * g.w + bb.w;
    if (NLN == 2) {
        s1[tid] = y0 + y1 + y2 + y3;
        s2[tid] = y0 * y0 + y1 * y1 + y2 * y2 + y3 * y3;
        __syncthreads();
        for (int st = 128; st > 0; st >>= 1) {
            if (tid < st) { s1[tid] += s1[tid + st]; s2[tid] += s2[tid + st]; }
            __syncthreads();
        }
        float mu2 = s1[0] * (1.f / DM);
        float var2 = s2[0] * (1.f / DM) - mu2 * mu2;
        float rs2 = rsqrtf(var2 + 1e-5f);
        float4 gg = reinterpret_cast<const float4*>(g2)[tid];
        float4 b2v = reinterpret_cast<const float4*>(b2)[tid];
        y0 = (y0 - mu2) * rs2 * gg.x + b2v.x;
        y1 = (y1 - mu2) * rs2 * gg.y + b2v.y;
        y2 = (y2 - mu2) * rs2 * gg.z + b2v.z;
        y3 = (y3 - mu2) * rs2 * gg.w + b2v.w;
    }
    unsigned short* o = out + (size_t)row * DM + tid * 4;
    o[0] = f2bf(y0); o[1] = f2bf(y1); o[2] = f2bf(y2); o[3] = f2bf(y3);
}

// ---------------------------------------------------------------------------
// bf16 WMMA GEMM:  out[M,N] = act(A[M,K] @ W[K,N] + bias) (+ res)
// Tile BM=128 BN=64 BK=32; 8 waves = 4(M) x 2(N), 2x2 WMMA tiles per wave.
// A tiles staged by TDM (tensor_load_to_lds) double-buffered and overlapped
// with compute via TENSORcnt; B tiles software-pipelined through VGPRs.
// ACT: 0=none 1=silu 2=softplus 3=gelu(exact)
// ---------------------------------------------------------------------------
#define GBM 128
#define GBN 64
#define GBK 32

template <int ACT, bool OUTBF, bool RES>
__global__ __launch_bounds__(256) void gemm_bf16_kernel(
    const unsigned short* __restrict__ A, const unsigned short* __restrict__ W,
    const float* __restrict__ bias, const float* __restrict__ res,
    float* __restrict__ outF, unsigned short* __restrict__ outB,
    int M, int N, int K) {
    __shared__ unsigned short sA[2][GBM * GBK];   // [m][k]  2 x 8 KB
    __shared__ unsigned short sB[2][GBN * GBK];   // [n][k]  2 x 4 KB (transposed)

    const int tid = threadIdx.x;
    const int lane = tid & 31, w = tid >> 5;
    const int wm = w & 3, wn = w >> 2;
    const int m0 = blockIdx.y * GBM;
    const int n0 = blockIdx.x * GBN;
    const int NT = K / GBK;

    v8f acc[2][2] = {};

    const int brow = tid >> 3, bc8 = (tid & 7) * 8;  // B tile: 32 rows x 8 cols
    const int mlan = lane & 15, kh = lane >> 4;

#ifdef USE_TDM
    unsigned ldsA[2] = { (unsigned)(uintptr_t)&sA[0][0],
                         (unsigned)(uintptr_t)&sA[1][0] };
    if (tid < 32)   // wave 0 drives the DMA engine
        tdm_load_tile_bf16(A + (size_t)m0 * K, ldsA[0],
                           (unsigned)K, (unsigned)M, GBK, GBM, (unsigned)K);
#else
    const int arow = tid >> 1, aseg = tid & 1;
    {   // cooperative stage of A tile 0
        const uint4* ag = reinterpret_cast<const uint4*>(
            A + (size_t)(m0 + arow) * K + aseg * 16);
        uint4 a0 = ag[0], a1 = ag[1];
        *reinterpret_cast<uint4*>(&sA[0][arow * GBK + aseg * 16]) = a0;
        *reinterpret_cast<uint4*>(&sA[0][arow * GBK + aseg * 16 + 8]) = a1;
    }
#endif
    {   // stage B tile 0 (transposed)
        uint4 bv = *reinterpret_cast<const uint4*>(
            W + (size_t)brow * N + n0 + bc8);
        const unsigned short* bs = reinterpret_cast<const unsigned short*>(&bv);
#pragma unroll
        for (int j = 0; j < 8; ++j) sB[0][(bc8 + j) * GBK + brow] = bs[j];
    }

    for (int it = 0; it < NT; ++it) {
        const int cur = it & 1, nxt = cur ^ 1;
        const bool hasNext = (it + 1 < NT);
        uint4 bv;
#ifndef USE_TDM
        uint4 av0, av1;
#endif
        if (hasNext) {
            const int kn = (it + 1) * GBK;
            bv = *reinterpret_cast<const uint4*>(
                W + (size_t)(kn + brow) * N + n0 + bc8);
#ifdef USE_TDM
            if (tid < 32)
                tdm_load_tile_bf16(A + (size_t)m0 * K + kn, ldsA[nxt],
                                   (unsigned)K, (unsigned)M, GBK, GBM,
                                   (unsigned)K);
#else
            const uint4* ag = reinterpret_cast<const uint4*>(
                A + (size_t)(m0 + arow) * K + kn + aseg * 16);
            av0 = ag[0]; av1 = ag[1];
#endif
        }
#ifdef USE_TDM
        if (tid < 32) {   // allow the next DMA to stay in flight
            if (hasNext) __builtin_amdgcn_s_wait_tensorcnt(1);
            else         __builtin_amdgcn_s_wait_tensorcnt(0);
        }
#endif
        __syncthreads();   // current A (TDM) + B (prev iter stores) visible

        union UF { uint4 q[2]; v16bf v; } aF[2], bF[2];
#pragma unroll
        for (int mi = 0; mi < 2; ++mi) {
            int r = wm * 32 + mi * 16 + mlan;
            aF[mi].q[0] = *reinterpret_cast<const uint4*>(&sA[cur][r * GBK + kh * 8]);
            aF[mi].q[1] = *reinterpret_cast<const uint4*>(&sA[cur][r * GBK + kh * 8 + 16]);
        }
#pragma unroll
        for (int ni = 0; ni < 2; ++ni) {
            int c = wn * 32 + ni * 16 + mlan;
            bF[ni].q[0] = *reinterpret_cast<const uint4*>(&sB[cur][c * GBK + kh * 8]);
            bF[ni].q[1] = *reinterpret_cast<const uint4*>(&sB[cur][c * GBK + kh * 8 + 16]);
        }
#pragma unroll
        for (int mi = 0; mi < 2; ++mi)
#pragma unroll
            for (int ni = 0; ni < 2; ++ni)
                acc[mi][ni] = __builtin_amdgcn_wmma_f32_16x16x32_bf16(
                    false, aF[mi].v, false, bF[ni].v,
                    (short)0, acc[mi][ni], false, false);

        if (hasNext) {   // publish next B tile after compute (latency hidden)
            const unsigned short* bs = reinterpret_cast<const unsigned short*>(&bv);
#pragma unroll
            for (int j = 0; j < 8; ++j) sB[nxt][(bc8 + j) * GBK + brow] = bs[j];
#ifndef USE_TDM
            *reinterpret_cast<uint4*>(&sA[nxt][arow * GBK + aseg * 16]) = av0;
            *reinterpret_cast<uint4*>(&sA[nxt][arow * GBK + aseg * 16 + 8]) = av1;
#endif
        }
        __syncthreads();   // readers done before next DMA/stores overwrite
    }

    // --- epilogue ---
    const int hi = lane >> 4;
#pragma unroll
    for (int mi = 0; mi < 2; ++mi)
#pragma unroll
        for (int ni = 0; ni < 2; ++ni) {
            int gcol = n0 + wn * 32 + ni * 16 + mlan;
            float bvv = bias[gcol];
#pragma unroll
            for (int r = 0; r < 8; ++r) {
                int grow = m0 + wm * 32 + mi * 16 + hi * 8 + r;
                float vv = acc[mi][ni][r] + bvv;
                if (ACT == 1) vv = vv / (1.f + __expf(-vv));
                else if (ACT == 2) vv = (vv > 20.f) ? vv : log1pf(__expf(vv));
                else if (ACT == 3) vv = 0.5f * vv * (1.f + erff(vv * 0.70710678118f));
                if (RES) vv += res[(size_t)grow * N + gcol];
                if (OUTBF) outB[(size_t)grow * N + gcol] = f2bf(vv);
                else       outF[(size_t)grow * N + gcol] = vv;
            }
        }
}

// ---------------------------------------------------------------------------
// Depthwise conv3 (pad 1) + SiLU. One thread per (b,l,d) element.
// ---------------------------------------------------------------------------
__global__ __launch_bounds__(256) void conv_silu_kernel(
    const float* __restrict__ x1, const float* __restrict__ cw,
    float* __restrict__ xc, unsigned short* __restrict__ xcb) {
    size_t i = (size_t)blockIdx.x * 256 + threadIdx.x;   // < TT*DM
    int d = (int)(i & (DM - 1));
    size_t bl = i >> 10;
    int l = (int)(bl & (LL - 1));
    float w0 = cw[d * 3 + 0], w1 = cw[d * 3 + 1], w2 = cw[d * 3 + 2];
    float xm1 = (l > 0)      ? x1[i - DM] : 0.f;
    float xp1 = (l < LL - 1) ? x1[i + DM] : 0.f;
    float v = xm1 * w0 + x1[i] * w1 + xp1 * w2;
    v = v / (1.f + __expf(-v));
    xc[i] = v;
    xcb[i] = f2bf(v);
}

// ---------------------------------------------------------------------------
// Selective scan: 65536 lanes, one per (b,d,n). 16-lane shuffle reduction.
// grid = B * (D/16) = 256 blocks x 256 threads; tid = dlocal*16 + n.
// ---------------------------------------------------------------------------
__global__ __launch_bounds__(256) void scan_kernel(
    const float* __restrict__ xc, const float* __restrict__ delta,
    const float* __restrict__ bc, const float* __restrict__ a_log,
    const float* __restrict__ dp, float* __restrict__ y) {
    const int tid = threadIdx.x;
    const int n = tid & 15;
    const int dl = tid >> 4;
    const int b = blockIdx.x >> 6;
    const int d = (blockIdx.x & 63) * 16 + dl;
    const float An = -__expf(a_log[d * NS + n]);
    const float Dp = dp[d];
    const float* xcb = xc + (size_t)b * LL * DM;
    const float* dtb = delta + (size_t)b * LL * DM;
    const float* bcb = bc + (size_t)b * LL * 64;
    float* yb = y + (size_t)b * LL * DM;
    float h = 0.f;
    for (int l = 0; l < LL; ++l) {
        float xv = xcb[(size_t)l * DM + d];
        float dt = dtb[(size_t)l * DM + d];
        float bt = bcb[l * 64 + n];
        float ct = bcb[l * 64 + 16 + n];
        float dA = __expf(dt * An);
        h = dA * h + dt * bt * xv;
        float p = h * ct;
        p += __shfl_xor(p, 1, 16);
        p += __shfl_xor(p, 2, 16);
        p += __shfl_xor(p, 4, 16);
        p += __shfl_xor(p, 8, 16);
        if (n == 0) yb[(size_t)l * DM + d] = p + Dp * xv;
    }
}

// ---------------------------------------------------------------------------
// p = bf16(y * v)
// ---------------------------------------------------------------------------
__global__ __launch_bounds__(256) void mul_cvt_kernel(
    const float* __restrict__ y, const float* __restrict__ v,
    unsigned short* __restrict__ p, int n) {
    int i = blockIdx.x * 256 + threadIdx.x;
    if (i < n) p[i] = f2bf(y[i] * v[i]);
}

// ---------------------------------------------------------------------------
// Host-side orchestration
// ---------------------------------------------------------------------------
extern "C" void kernel_launch(void* const* d_in, const int* in_sizes, int n_in,
                              void* d_out, int out_size, void* d_ws, size_t ws_size,
                              hipStream_t stream) {
    const float* x      = (const float*)d_in[0];
    const float* ln1_g  = (const float*)d_in[1];
    const float* ln1_b  = (const float*)d_in[2];
    const float* ln_g   = (const float*)d_in[3];
    const float* ln_b   = (const float*)d_in[4];
    const float* w1_w   = (const float*)d_in[5];
    const float* w1_b   = (const float*)d_in[6];
    const float* v1_w   = (const float*)d_in[7];
    const float* v1_b   = (const float*)d_in[8];
    const float* w2_w   = (const float*)d_in[9];
    const float* w2_b   = (const float*)d_in[10];
    const float* conv_w = (const float*)d_in[11];
    const float* dt_w   = (const float*)d_in[12];
    const float* dt_b   = (const float*)d_in[13];
    const float* Bp_w   = (const float*)d_in[14];
    const float* Bp_b   = (const float*)d_in[15];
    const float* Cp_w   = (const float*)d_in[16];
    const float* Cp_b   = (const float*)d_in[17];
    const float* A_log  = (const float*)d_in[18];
    const float* D_p    = (const float*)d_in[19];
    const float* ln2_g  = (const float*)d_in[20];
    const float* ln2_b  = (const float*)d_in[21];
    const float* ff1_w  = (const float*)d_in[22];
    const float* ff1_b  = (const float*)d_in[23];
    const float* ff2_w  = (const float*)d_in[24];
    const float* ff2_b  = (const float*)d_in[25];

    // ---- workspace carve-up ----
    char* p = (char*)d_ws;
    const size_t fTD = (size_t)TT * DM * sizeof(float);        // 32 MB
    float* x1buf   = (float*)p;            p += fTD;           // x1 -> y
    float* vbuf    = (float*)p;            p += fTD;           // v
    float* xcbuf   = (float*)p;            p += fTD;           // xc -> x2
    float* dltbuf  = (float*)p;            p += fTD;           // delta
    float* bcbuf   = (float*)p;            p += (size_t)TT * 64 * sizeof(float);
    unsigned short* g1 = (unsigned short*)p; p += (size_t)TT * DM * 2;   // bf16 acts
    unsigned short* g2 = (unsigned short*)p; p += (size_t)TT * DFF * 2;  // gelu out
    unsigned short* wbuf = (unsigned short*)p; p += (size_t)DM * DFF * 2; // weights
    unsigned short* wbc  = (unsigned short*)p; p += (size_t)DM * 64 * 2;
    float* bcat = (float*)p;               p += 64 * sizeof(float);

    float* outF = (float*)d_out;

    const int cvtDD  = (DM * DM + 255) / 256;
    const int cvtFF  = (DM * DFF + 255) / 256;
    const int ewTD   = (TT * DM + 255) / 256;
    dim3 blk(256);

    // 1) fused LN1 + LN  ->  xm (bf16) in g1
    ln_kernel<2><<<TT, blk, 0, stream>>>(x, ln1_g, ln1_b, ln_g, ln_b, g1);

    // 2) x1 = xm @ w1 + b
    cvt_bf16_kernel<<<cvtDD, blk, 0, stream>>>(w1_w, wbuf, DM * DM);
    gemm_bf16_kernel<0, false, false><<<dim3(DM / GBN, TT / GBM), blk, 0, stream>>>(
        g1, wbuf, w1_b, nullptr, x1buf, nullptr, TT, DM, DM);

    // 3) v = silu(xm @ v1 + b)
    cvt_bf16_kernel<<<cvtDD, blk, 0, stream>>>(v1_w, wbuf, DM * DM);
    gemm_bf16_kernel<1, false, false><<<dim3(DM / GBN, TT / GBM), blk, 0, stream>>>(
        g1, wbuf, v1_b, nullptr, vbuf, nullptr, TT, DM, DM);

    // 4) xc = silu(conv3(x1))   (also bf16 into g1)
    conv_silu_kernel<<<ewTD, blk, 0, stream>>>(x1buf, conv_w, xcbuf, g1);

    // 5) delta = softplus(xc @ dt_w + dt_b)
    cvt_bf16_kernel<<<cvtDD, blk, 0, stream>>>(dt_w, wbuf, DM * DM);
    gemm_bf16_kernel<2, false, false><<<dim3(DM / GBN, TT / GBM), blk, 0, stream>>>(
        g1, wbuf, dt_b, nullptr, dltbuf, nullptr, TT, DM, DM);

    // 6) [Bt|Ct] = xc @ [Bp|Cp] + bias   (N padded to 64)
    pack_bc_kernel<<<(DM * 64 + 255) / 256, blk, 0, stream>>>(
        Bp_w, Cp_w, Bp_b, Cp_b, wbc, bcat);
    gemm_bf16_kernel<0, false, false><<<dim3(1, TT / GBM), blk, 0, stream>>>(
        g1, wbc, bcat, nullptr, bcbuf, nullptr, TT, 64, DM);

    // 7) selective scan -> y (reuses x1buf)
    scan_kernel<<<BB * (DM / 16), blk, 0, stream>>>(
        xcbuf, dltbuf, bcbuf, A_log, D_p, x1buf);

    // 8) p = bf16(y * v)
    mul_cvt_kernel<<<ewTD, blk, 0, stream>>>(x1buf, vbuf, g1, TT * DM);

    // 9) x2 = p @ w2 + b + x   (residual) -> xcbuf
    cvt_bf16_kernel<<<cvtDD, blk, 0, stream>>>(w2_w, wbuf, DM * DM);
    gemm_bf16_kernel<0, false, true><<<dim3(DM / GBN, TT / GBM), blk, 0, stream>>>(
        g1, wbuf, w2_b, x, xcbuf, nullptr, TT, DM, DM);

    // 10) xn2 = LN(x2) -> bf16 g1
    ln_kernel<1><<<TT, blk, 0, stream>>>(xcbuf, ln2_g, ln2_b, nullptr, nullptr, g1);

    // 11) h = gelu(xn2 @ ff1 + b) -> bf16 g2
    cvt_bf16_kernel<<<cvtFF, blk, 0, stream>>>(ff1_w, wbuf, DM * DFF);
    gemm_bf16_kernel<3, true, false><<<dim3(DFF / GBN, TT / GBM), blk, 0, stream>>>(
        g1, wbuf, ff1_b, nullptr, nullptr, g2, TT, DFF, DM);

    // 12) out = h @ ff2 + b + x2
    cvt_bf16_kernel<<<cvtFF, blk, 0, stream>>>(ff2_w, wbuf, DM * DFF);
    gemm_bf16_kernel<0, false, true><<<dim3(DM / GBN, TT / GBM), blk, 0, stream>>>(
        g2, wbuf, ff2_b, xcbuf, outF, nullptr, TT, DM, DFF);

    (void)in_sizes; (void)n_in; (void)out_size; (void)ws_size;
}
